// Gate_13864154432371
// MI455X (gfx1250) — compile-verified, matrix operand order checked by reading, not run on hardware
//
#include <hip/hip_runtime.h>
#include <hip/hip_bf16.h>

typedef __attribute__((ext_vector_type(16))) __bf16 v16bf;
typedef __attribute__((ext_vector_type(8)))  float  v8f;

#define DIM   7168
#define NEXP  256
#define NTOK  8192
#define TOPK  8
#define MB    32            // tokens per block
#define KTILES (DIM / 32)   // 224 k-steps

// LDS layout (byte offsets inside smem)
#define A_OFF(buf)  ((buf) * 2304)             // 3 x [32][36] bf16
#define B_OFF(buf)  (6912 + (buf) * 16384)     // 3 x 16 KB fragment blocks
#define BIAS_OFF    56064
#define SMEM_BYTES  57088

// ---------------------------------------------------------------------------
// f32 -> bf16 (round-to-nearest-even) raw bits
// ---------------------------------------------------------------------------
__device__ __forceinline__ unsigned f32_to_bf16_bits(float f) {
    unsigned u = __builtin_bit_cast(unsigned, f);
    unsigned rounded = u + 0x7fffu + ((u >> 16) & 1u);
    return rounded >> 16;
}

// CDNA5 async copy: global memory -> LDS, 16B per lane, tracked by ASYNCcnt.
__device__ __forceinline__ void async_b128(unsigned lds_off, unsigned long long gaddr) {
    asm volatile("global_load_async_to_lds_b128 %0, %1, off"
                 :: "v"(lds_off), "v"(gaddr)
                 : "memory");
}

// ---------------------------------------------------------------------------
// Repack weight [256 x 7168] f32 -> bf16, pre-swizzled into WMMA B-fragment
// order (32K x 16N fragments; each lane's 16 bf16 contiguous), kt-major so the
// GEMM streams one contiguous 16 KB block per k-step.
//   element index: gid = (((kt*16 + nt)*32 + L)*16 + e)
// ---------------------------------------------------------------------------
__global__ __launch_bounds__(256) void repack_weight(const float* __restrict__ w,
                                                     unsigned short* __restrict__ wp) {
    int gid  = blockIdx.x * 256 + threadIdx.x;
    int e    = gid & 15;
    int L    = (gid >> 4) & 31;
    int frag = gid >> 9;
    int nt   = frag & 15;
    int kt   = frag >> 4;

    int j    = e >> 1, p = e & 1;
    int g    = j >> 2, jm = j & 3;
    int half = L >> 4, nl = L & 15;
    int kl   = (g << 4) + (half << 3) + (jm << 1) + p;  // 0..31
    int k    = kt * 32 + kl;
    int n    = nt * 16 + nl;

    wp[gid] = (unsigned short)f32_to_bf16_bits(w[(size_t)n * DIM + k]);
}

// ---------------------------------------------------------------------------
// Fused gate kernel: bf16 WMMA GEMM (32 tokens x 256 experts, K=7168) with
// async-to-LDS triple-buffered pipeline + sigmoid + group-limited top-k.
// ---------------------------------------------------------------------------
__global__ __launch_bounds__(256, 2) void gate_kernel(const float* __restrict__ x,
                                                      const float* __restrict__ bias,
                                                      const unsigned short* __restrict__ wp,
                                                      float* __restrict__ out_w,
                                                      int*   __restrict__ out_i) {
    __shared__ __align__(16) char smem[SMEM_BYTES];
    float* scores   = (float*)smem;              // [32][257], overlaps dead GEMM buffers
    float* bias_lds = (float*)(smem + BIAS_OFF);
    const unsigned smem_base = (unsigned)(size_t)smem;  // LDS byte offset of smem

    const int tid  = threadIdx.x;
    const int w    = tid >> 5;   // wave 0..7
    const int L    = tid & 31;   // lane
    const int half = L >> 4;
    const int t0   = blockIdx.x * MB;

    bias_lds[tid] = bias[tid];

    // A-staging coordinates: thread -> (row r of 32, float4 chunk c of 8)
    const int r = tid >> 3, c = tid & 7;
    const float* xrow = x + (size_t)(t0 + r) * DIM + c * 4;

    // issue one B tile (16 KB) as 4 async b128 per thread
    auto issueB = [&](int tile, int buf) {
        unsigned long long g = (unsigned long long)(size_t)(wp + (size_t)tile * 8192);
        unsigned lds = smem_base + B_OFF(buf);
        #pragma unroll
        for (int q = 0; q < 4; ++q)
            async_b128(lds + (unsigned)(tid + 256 * q) * 16,
                       g + (unsigned long long)(tid + 256 * q) * 16);
    };
    auto stageA = [&](int buf, float4 va) {
        unsigned short* An = (unsigned short*)(smem + A_OFF(buf));
        uint2 pk;
        pk.x = f32_to_bf16_bits(va.x) | (f32_to_bf16_bits(va.y) << 16);
        pk.y = f32_to_bf16_bits(va.z) | (f32_to_bf16_bits(va.w) << 16);
        *(uint2*)(An + r * 36 + c * 4) = pk;    // 8B-aligned, conflict-free stride 36
    };

    // ---- prologue: tiles 0 and 1 in flight ----
    issueB(0, 0);
    issueB(1, 1);
    float4 a_reg = *(const float4*)(xrow);           // A tile 0
    stageA(0, a_reg);
    a_reg = *(const float4*)(xrow + 32);             // A tile 1 (held in regs)

    v8f acc[4];
    #pragma unroll
    for (int i = 0; i < 4; ++i)
        #pragma unroll
        for (int j = 0; j < 8; ++j) acc[i][j] = 0.0f;

    const int mrow   = (w >> 2) << 4;  // 0 or 16
    const int ntbase = (w & 3) * 4;    // 4 N-tiles of 16 experts per wave
    const int m      = mrow + (L & 15);

    int bcur = 0;  // kt % 3
    for (int kt = 0; kt < KTILES; ++kt) {
        int bnxt = bcur + 1; if (bnxt == 3) bnxt = 0;   // (kt+1) % 3
        int bnn  = bnxt + 1; if (bnn  == 3) bnn  = 0;   // (kt+2) % 3

        // own async portion of tile kt complete (in-order: <=4 leaves only
        // tile kt+1's 4 instructions outstanding); barrier publishes all waves'
        // portions and closes out iteration kt-1's LDS readers.
        if (kt + 1 < KTILES) asm volatile("s_wait_asynccnt 0x4" ::: "memory");
        else                 asm volatile("s_wait_asynccnt 0x0" ::: "memory");
        __syncthreads();

        if (kt + 1 < KTILES) stageA(bnxt, a_reg);                         // A tile kt+1 -> LDS
        if (kt + 2 < KTILES) {
            a_reg = *(const float4*)(xrow + (size_t)(kt + 2) * 32);       // prefetch A kt+2
            issueB(kt + 2, bnn);                                          // async B kt+2
        }

        // ---- compute tile kt ----
        const unsigned short* Ac = (const unsigned short*)(smem + A_OFF(bcur));
        const char*           Bc = smem + B_OFF(bcur);

        union { unsigned u[8]; v16bf v; } af;
        #pragma unroll
        for (int jj = 0; jj < 8; ++jj) {
            int kb = (jj < 4) ? (half * 8 + jj * 2) : (16 + half * 8 + (jj - 4) * 2);
            af.u[jj] = *(const unsigned*)(Ac + m * 36 + kb);
        }
        #pragma unroll
        for (int t = 0; t < 4; ++t) {
            int nt = ntbase + t;
            union { uint4 q[2]; v16bf v; } bf;
            const uint4* bp = (const uint4*)(Bc + nt * 1024 + L * 32);
            bf.q[0] = bp[0];
            bf.q[1] = bp[1];
            acc[t] = __builtin_amdgcn_wmma_f32_16x16x32_bf16(
                false, af.v, false, bf.v, (short)0, acc[t], false, false);
        }
        bcur = bnxt;
    }

    __syncthreads();   // scores region overlaps GEMM buffers: wait for all compute

    // ---- sigmoid scores -> LDS [32][257] (fast rcp) ----
    #pragma unroll
    for (int t = 0; t < 4; ++t) {
        int n = (ntbase + t) * 16 + (L & 15);
        #pragma unroll
        for (int v = 0; v < 8; ++v) {
            int mm = mrow + v + 8 * half;
            scores[mm * 257 + n] = __builtin_amdgcn_rcpf(1.0f + __expf(-acc[t][v]));
        }
    }
    __syncthreads();

    // ---- routing: one lane per token ----
    if (tid < MB) {
        const float* sc = scores + tid * 257;

        float gs[8];
        #pragma unroll
        for (int g = 0; g < 8; ++g) {
            float m1 = -1e30f, m2 = -1e30f;
            for (int e = 0; e < 32; ++e) {
                float v = sc[g * 32 + e] + bias_lds[g * 32 + e];
                if (v > m1)      { m2 = m1; m1 = v; }
                else if (v > m2) { m2 = v; }
            }
            gs[g] = m1 + m2;
        }
        unsigned gmask = 0;
        for (int i = 0; i < 4; ++i) {
            float best = -1e30f; int bg = 0;
            for (int g = 0; g < 8; ++g)
                if (!((gmask >> g) & 1) && gs[g] > best) { best = gs[g]; bg = g; }
            gmask |= 1u << bg;
        }
        unsigned chosen[8] = {0, 0, 0, 0, 0, 0, 0, 0};
        int   eidx[TOPK];
        float ew[TOPK];
        float esum = 0.0f;
        for (int i = 0; i < TOPK; ++i) {
            float best = -1e30f; int be = 0;
            for (int e = 0; e < NEXP; ++e) {
                if ((chosen[e >> 5] >> (e & 31)) & 1) continue;
                float v = ((gmask >> (e >> 5)) & 1) ? (sc[e] + bias_lds[e]) : 0.0f;
                if (v > best) { best = v; be = e; }
            }
            chosen[be >> 5] |= 1u << (be & 31);
            eidx[i] = be;
            float wv = sc[be];          // weights from ORIGINAL sigmoid scores
            ew[i] = wv;
            esum += wv;
        }
        const float scale = 2.5f / esum;
        const int tok = t0 + tid;
        #pragma unroll
        for (int i = 0; i < TOPK; ++i) {
            out_w[tok * TOPK + i] = ew[i] * scale;
            out_i[tok * TOPK + i] = eidx[i];
        }
    }
}

// ---------------------------------------------------------------------------
extern "C" void kernel_launch(void* const* d_in, const int* in_sizes, int n_in,
                              void* d_out, int out_size, void* d_ws, size_t ws_size,
                              hipStream_t stream) {
    const float* x      = (const float*)d_in[0];
    // d_in[1] token_mask: unused by the reference
    const float* weight = (const float*)d_in[2];
    const float* bias   = (const float*)d_in[3];

    unsigned short* wp = (unsigned short*)d_ws;  // 7168*256 bf16 = 3.67 MB

    repack_weight<<<(DIM * NEXP) / 256, 256, 0, stream>>>(weight, wp);

    float* out_w = (float*)d_out;
    int*   out_i = (int*)d_out + (size_t)NTOK * TOPK;
    gate_kernel<<<NTOK / MB, 256, 0, stream>>>(x, bias, wp, out_w, out_i);
}
